// SchNetModel_29454885716581
// MI455X (gfx1250) — matrix operand order repaired and checked by reading, not compiled, not run on hardware
//
#include <hip/hip_runtime.h>
#include <math.h>

#define NATOMS 100000
#define NEDGES 3200000
#define NMOL   2000
#define HID    64
#define NGAUSS 50

typedef _Float16 v16h __attribute__((ext_vector_type(16)));
typedef float    v8f  __attribute__((ext_vector_type(8)));

#define GDELTA  (10.0f/49.0f)
#define GCOEFF  (-0.5f/(GDELTA*GDELTA))
#define PI_OC   0.31415926535897931f   /* pi / cutoff */
#define LOG2F_  0.69314718055994531f

__device__ __forceinline__ float sspf(float x) {
  // shifted softplus, numerically stable
  return fmaxf(x, 0.f) + log1pf(__expf(-fabsf(x))) - LOG2F_;
}

__device__ __forceinline__ v8f wmma16(v16h a, v16h b, v8f c) {
  return __builtin_amdgcn_wmma_f32_16x16x32_f16(false, a, false, b, (short)0, c, false, false);
}

__device__ __forceinline__ v8f zero8() {
  v8f z;
#pragma unroll
  for (int i = 0; i < 8; ++i) z[i] = 0.f;
  return z;
}

// A-matrix (16x32 f16) K index for vector element e, half-wave hi
__device__ __forceinline__ int akm(int e, int hi) {
  return e + hi * 16;
}

__device__ __forceinline__ void atomAddF(float* p, float v) {
  __hip_atomic_fetch_add(p, v, __ATOMIC_RELAXED, __HIP_MEMORY_SCOPE_AGENT);
}

// Same-wave LDS RAW sync: all prior ds ops complete. Cheaper than
// __syncthreads() (no workgroup fence -> no load/storecnt stall on the
// in-flight scatter atomics). Valid because blocks are exactly one wave
// and the LDS pipe processes a wave's ds ops in order.
__device__ __forceinline__ void lds_sync() {
  asm volatile("s_wait_dscnt 0" ::: "memory");
}
// Compiler-only reorder barrier (WAR through LDS is in-order in hardware).
__device__ __forceinline__ void mem_order() {
  asm volatile("" ::: "memory");
}

// ---------------------------------------------------------------- embedding
__global__ __launch_bounds__(256) void embed_kernel(
    const int* __restrict__ z, const float* __restrict__ emb,
    float* __restrict__ h, int total) {
  int i = blockIdx.x * 256 + threadIdx.x;
  if (i >= total) return;
  int n = i >> 6, c = i & 63;
  h[i] = emb[z[n] * HID + c];
}

// ---------------------------------------------------------------- distances
__global__ __launch_bounds__(256) void dist_kernel(
    const float* __restrict__ pos, const int* __restrict__ esrc,
    const int* __restrict__ edst, float* __restrict__ dvec, int E) {
  int e = blockIdx.x * 256 + threadIdx.x;
  if (e >= E) return;
  int s = esrc[e], t = edst[e];
  float dx = pos[s * 3 + 0] - pos[t * 3 + 0];
  float dy = pos[s * 3 + 1] - pos[t * 3 + 1];
  float dz = pos[s * 3 + 2] - pos[t * 3 + 2];
  dvec[e] = sqrtf(dx * dx + dy * dy + dz * dz);
}

// ------------------------------------------------ x = h @ cf_lin1, agg = 0
__global__ __launch_bounds__(32) void node_pre_kernel(
    const float* __restrict__ h, const float* __restrict__ w,
    float* __restrict__ x, float* __restrict__ agg, int numTiles) {
  const int lane = threadIdx.x & 31, hi = lane >> 4, ln = lane & 15;
  v16h B[4][2];
#pragma unroll
  for (int t = 0; t < 4; ++t)
#pragma unroll
    for (int kc = 0; kc < 2; ++kc)
#pragma unroll
      for (int e = 0; e < 16; ++e) {
        const int K = e + hi * 16 + 32 * kc;
        B[t][kc][e] = (_Float16)w[K * HID + 16 * t + ln];
      }
  __shared__ float sH[16 * HID];
  for (int tile = blockIdx.x; tile < numTiles; tile += gridDim.x) {
    const int base = tile * 16;
    mem_order();
    for (int i = lane * 4; i < 16 * HID; i += 32 * 4)
      *(float4*)(sH + i) = *(const float4*)(h + (size_t)base * HID + i);
    lds_sync();
    v16h a[2];
#pragma unroll
    for (int kc = 0; kc < 2; ++kc)
#pragma unroll
      for (int e = 0; e < 16; ++e)
        a[kc][e] = (_Float16)sH[ln * HID + akm(e, hi) + 32 * kc];
    v8f acc[4];
#pragma unroll
    for (int t = 0; t < 4; ++t) {
      acc[t] = wmma16(a[0], B[t][0], zero8());
      acc[t] = wmma16(a[1], B[t][1], acc[t]);
    }
#pragma unroll
    for (int r = 0; r < 8; ++r) {
      const size_t rowoff = (size_t)(base + r + hi * 8) * HID + ln;
      float* xr = x + rowoff;
      float* ar = agg + rowoff;
#pragma unroll
      for (int t = 0; t < 4; ++t) {
        xr[16 * t] = acc[t][r];
        ar[16 * t] = 0.f;
      }
    }
  }
}

// -------- edge filter network + gather/scatter message passing (the hot loop)
__global__ __launch_bounds__(32) void edge_kernel(
    const float* __restrict__ dvec,
    const int* __restrict__ esrc, const int* __restrict__ edst,
    const float* __restrict__ xin, float* __restrict__ agg,
    const float* __restrict__ w1, const float* __restrict__ b1,
    const float* __restrict__ w2, const float* __restrict__ b2,
    int numTiles) {
  const int lane = threadIdx.x & 31, hi = lane >> 4, ln = lane & 15;

  // B matrices resident in registers in WMMA B-layout; reused across tiles
  v16h B1[4][2], B2[4][2];
  float b1c[4], b2c[4];
#pragma unroll
  for (int t = 0; t < 4; ++t) {
    const int col = 16 * t + ln;
#pragma unroll
    for (int kc = 0; kc < 2; ++kc)
#pragma unroll
      for (int e = 0; e < 16; ++e) {
        const int K = e + hi * 16 + 32 * kc;
        B1[t][kc][e] = (_Float16)((K < NGAUSS) ? w1[K * HID + col] : 0.f);
        B2[t][kc][e] = (_Float16)w2[K * HID + col];
      }
    b1c[t] = b1[col];
    b2c[t] = b2[col];
  }

  __shared__ _Float16 sMid[16 * 72];  // padded row stride vs bank conflicts

  for (int tile = blockIdx.x; tile < numTiles; tile += gridDim.x) {
    const int base = tile * 16;

    // RBF A-matrix built in-lane (lane owns row ln; K subset fixed by layout)
    const float dm = dvec[base + ln];
    v16h a[2];
#pragma unroll
    for (int kc = 0; kc < 2; ++kc)
#pragma unroll
      for (int e = 0; e < 16; ++e) {
        const int K = akm(e, hi) + 32 * kc;
        float v = 0.f;
        if (K < NGAUSS) {
          const float u = dm - (float)K * GDELTA;
          v = __expf(GCOEFF * u * u);
        }
        a[kc][e] = (_Float16)v;
      }

    // GEMM1: rbf[16x64] @ w1[64x64]
    v8f acc[4];
#pragma unroll
    for (int t = 0; t < 4; ++t) {
      acc[t] = wmma16(a[0], B1[t][0], zero8());
      acc[t] = wmma16(a[1], B1[t][1], acc[t]);
    }

    // bias + ssp, C-layout -> A-layout via LDS (same-wave, in-order DS pipe)
    mem_order();
#pragma unroll
    for (int t = 0; t < 4; ++t)
#pragma unroll
      for (int r = 0; r < 8; ++r)
        sMid[(r + hi * 8) * 72 + 16 * t + ln] = (_Float16)sspf(acc[t][r] + b1c[t]);
    lds_sync();
    v16h a2[2];
#pragma unroll
    for (int kc = 0; kc < 2; ++kc)
#pragma unroll
      for (int e = 0; e < 16; ++e)
        a2[kc][e] = sMid[ln * 72 + akm(e, hi) + 32 * kc];

    // GEMM2: mid[16x64] @ w2[64x64]
#pragma unroll
    for (int t = 0; t < 4; ++t) {
      acc[t] = wmma16(a2[0], B2[t][0], zero8());
      acc[t] = wmma16(a2[1], B2[t][1], acc[t]);
    }

    // cosine cutoff, gather x[src], scatter-add into agg[dst]
#pragma unroll
    for (int r = 0; r < 8; ++r) {
      const int row = r + hi * 8;
      const int e = base + row;
      const float Cc = 0.5f * (__cosf(dvec[e] * PI_OC) + 1.f);
      const int s = esrc[e], d = edst[e];
      const float* xr = xin + (size_t)s * HID + ln;
      float* ar = agg + (size_t)d * HID + ln;
#pragma unroll
      for (int t = 0; t < 4; ++t) {
        const float wv = (acc[t][r] + b2c[t]) * Cc;
        atomAddF(ar + 16 * t, wv * xr[16 * t]);
      }
    }
  }
}

// --------- h += int_lin( ssp( agg @ cf_lin2 + b ) ) + b  (two chained GEMMs)
__global__ __launch_bounds__(32) void node_post_kernel(
    float* __restrict__ h, const float* __restrict__ agg,
    const float* __restrict__ wA, const float* __restrict__ bA,
    const float* __restrict__ wB, const float* __restrict__ bB,
    int numTiles) {
  const int lane = threadIdx.x & 31, hi = lane >> 4, ln = lane & 15;
  v16h BA[4][2], BB[4][2];
  float bAc[4], bBc[4];
#pragma unroll
  for (int t = 0; t < 4; ++t) {
    const int col = 16 * t + ln;
#pragma unroll
    for (int kc = 0; kc < 2; ++kc)
#pragma unroll
      for (int e = 0; e < 16; ++e) {
        const int K = e + hi * 16 + 32 * kc;
        BA[t][kc][e] = (_Float16)wA[K * HID + col];
        BB[t][kc][e] = (_Float16)wB[K * HID + col];
      }
    bAc[t] = bA[col];
    bBc[t] = bB[col];
  }
  __shared__ float sIn[16 * HID];
  __shared__ _Float16 sMid[16 * 72];
  for (int tile = blockIdx.x; tile < numTiles; tile += gridDim.x) {
    const int base = tile * 16;
    mem_order();
    for (int i = lane * 4; i < 16 * HID; i += 32 * 4)
      *(float4*)(sIn + i) = *(const float4*)(agg + (size_t)base * HID + i);
    lds_sync();
    v16h a[2];
#pragma unroll
    for (int kc = 0; kc < 2; ++kc)
#pragma unroll
      for (int e = 0; e < 16; ++e)
        a[kc][e] = (_Float16)sIn[ln * HID + akm(e, hi) + 32 * kc];
    v8f acc[4];
#pragma unroll
    for (int t = 0; t < 4; ++t) {
      acc[t] = wmma16(a[0], BA[t][0], zero8());
      acc[t] = wmma16(a[1], BA[t][1], acc[t]);
    }
    mem_order();
#pragma unroll
    for (int t = 0; t < 4; ++t)
#pragma unroll
      for (int r = 0; r < 8; ++r)
        sMid[(r + hi * 8) * 72 + 16 * t + ln] = (_Float16)sspf(acc[t][r] + bAc[t]);
    lds_sync();
    v16h a2[2];
#pragma unroll
    for (int kc = 0; kc < 2; ++kc)
#pragma unroll
      for (int e = 0; e < 16; ++e)
        a2[kc][e] = sMid[ln * 72 + akm(e, hi) + 32 * kc];
#pragma unroll
    for (int t = 0; t < 4; ++t) {
      acc[t] = wmma16(a2[0], BB[t][0], zero8());
      acc[t] = wmma16(a2[1], BB[t][1], acc[t]);
    }
#pragma unroll
    for (int r = 0; r < 8; ++r) {
      float* hr = h + (size_t)(base + r + hi * 8) * HID + ln;
#pragma unroll
      for (int t = 0; t < 4; ++t)
        hr[16 * t] += acc[t][r] + bBc[t];
    }
  }
}

// ---------------------------------------------------------------- zero pool
__global__ __launch_bounds__(256) void zero_kernel(float* __restrict__ p, int n) {
  int i = blockIdx.x * 256 + threadIdx.x;
  if (i < n) p[i] = 0.f;
}

// --------------------------------- output head 64->32->1 + molecule pooling
__global__ __launch_bounds__(256) void head_kernel(
    const float* __restrict__ h, const int* __restrict__ batch,
    const float* __restrict__ w1, const float* __restrict__ b1,
    const float* __restrict__ w2, const float* __restrict__ b2,
    float* __restrict__ pool, int N) {
  __shared__ float sw1[HID * 32];
  __shared__ float sb1[32], sw2[32], sb2;
  for (int i = threadIdx.x; i < HID * 32; i += 256) sw1[i] = w1[i];
  if (threadIdx.x < 32) {
    sb1[threadIdx.x] = b1[threadIdx.x];
    sw2[threadIdx.x] = w2[threadIdx.x];
  }
  if (threadIdx.x == 0) sb2 = b2[0];
  __syncthreads();
  int n = blockIdx.x * 256 + threadIdx.x;
  if (n >= N) return;
  float t[32];
#pragma unroll
  for (int j = 0; j < 32; ++j) t[j] = sb1[j];
  const float4* hp = (const float4*)(h + (size_t)n * HID);
#pragma unroll
  for (int k4 = 0; k4 < HID / 4; ++k4) {
    const float4 hv = hp[k4];
    const int k = k4 * 4;
#pragma unroll
    for (int j = 0; j < 32; ++j) {
      t[j] += hv.x * sw1[(k + 0) * 32 + j];
      t[j] += hv.y * sw1[(k + 1) * 32 + j];
      t[j] += hv.z * sw1[(k + 2) * 32 + j];
      t[j] += hv.w * sw1[(k + 3) * 32 + j];
    }
  }
  float val = sb2;
#pragma unroll
  for (int j = 0; j < 32; ++j) val += sspf(t[j]) * sw2[j];
  atomAddF(&pool[batch[n]], val);
}

// ------------------------------------------------------- final Linear(1,1)
__global__ __launch_bounds__(256) void final_kernel(
    const float* __restrict__ pool, const float* __restrict__ fw,
    const float* __restrict__ fb, float* __restrict__ out, int G) {
  int g = blockIdx.x * 256 + threadIdx.x;
  if (g < G) out[g] = pool[g] * fw[0] + fb[0];
}

extern "C" void kernel_launch(void* const* d_in, const int* in_sizes, int n_in,
                              void* d_out, int out_size, void* d_ws, size_t ws_size,
                              hipStream_t stream) {
  (void)in_sizes; (void)n_in; (void)out_size; (void)ws_size;
  const int*   z       = (const int*)d_in[0];
  const float* pos     = (const float*)d_in[1];
  const int*   batch   = (const int*)d_in[2];
  const int*   ei      = (const int*)d_in[3];     // [2, E]
  const float* emb     = (const float*)d_in[4];
  const float* ow1     = (const float*)d_in[5];
  const float* ob1     = (const float*)d_in[6];
  const float* ow2     = (const float*)d_in[7];
  const float* ob2     = (const float*)d_in[8];
  const float* fw      = (const float*)d_in[9];
  const float* fb      = (const float*)d_in[10];
  const float* mw1     = (const float*)d_in[11];  // [3,50,64]
  const float* mb1     = (const float*)d_in[12];
  const float* mw2     = (const float*)d_in[13];  // [3,64,64]
  const float* mb2     = (const float*)d_in[14];
  const float* cf1w    = (const float*)d_in[15];
  const float* cf2w    = (const float*)d_in[16];
  const float* cf2b    = (const float*)d_in[17];
  const float* ilw     = (const float*)d_in[18];
  const float* ilb     = (const float*)d_in[19];

  char* ws = (char*)d_ws;
  float* h    = (float*)(ws);                         // N*64 f32 = 25.6 MB
  float* x    = (float*)(ws + (size_t)25600000);      // N*64
  float* agg  = (float*)(ws + (size_t)51200000);      // N*64
  float* dvec = (float*)(ws + (size_t)76800000);      // E f32 = 12.8 MB
  float* pool = (float*)(ws + (size_t)89600000);      // G f32

  embed_kernel<<<(NATOMS * HID + 255) / 256, 256, 0, stream>>>(z, emb, h, NATOMS * HID);
  dist_kernel<<<(NEDGES + 255) / 256, 256, 0, stream>>>(pos, ei, ei + NEDGES, dvec, NEDGES);

  for (int l = 0; l < 3; ++l) {
    node_pre_kernel<<<4096, 32, 0, stream>>>(h, cf1w + (size_t)l * HID * HID, x, agg, NATOMS / 16);
    edge_kernel<<<16384, 32, 0, stream>>>(dvec, ei, ei + NEDGES, x, agg,
                                          mw1 + (size_t)l * NGAUSS * HID, mb1 + (size_t)l * HID,
                                          mw2 + (size_t)l * HID * HID,   mb2 + (size_t)l * HID,
                                          NEDGES / 16);
    node_post_kernel<<<4096, 32, 0, stream>>>(h, agg,
                                              cf2w + (size_t)l * HID * HID, cf2b + (size_t)l * HID,
                                              ilw + (size_t)l * HID * HID,  ilb + (size_t)l * HID,
                                              NATOMS / 16);
  }

  zero_kernel<<<(NMOL + 255) / 256, 256, 0, stream>>>(pool, NMOL);
  head_kernel<<<(NATOMS + 255) / 256, 256, 0, stream>>>(h, batch, ow1, ob1, ow2, ob2, pool, NATOMS);
  final_kernel<<<(NMOL + 255) / 256, 256, 0, stream>>>(pool, fw, fb, (float*)d_out, NMOL);
}
